// GIN_29850022707973
// MI455X (gfx1250) — matrix operand order repaired and checked by reading, not compiled, not run on hardware
//
#include <hip/hip_runtime.h>
#include <hip/hip_bf16.h>

#define NNODES   50000
#define NEDGES   800000
#define IN_DIM   128
#define HID      256
#define OUT_DIM  128
#define EDGE_DIM 32

typedef __attribute__((ext_vector_type(16))) __bf16 v16bf;
typedef __attribute__((ext_vector_type(8)))  __bf16 v8bf;
typedef __attribute__((ext_vector_type(8)))  float  v8f;

__device__ __forceinline__ __bf16 f2bf(float f) {
  unsigned u = __builtin_bit_cast(unsigned, f);
  u += 0x7FFFu + ((u >> 16) & 1u);               // round-to-nearest-even
  return __builtin_bit_cast(__bf16, (unsigned short)(u >> 16));
}

// wT[c][k] = bf16(w[k][c]) ; w is K x C row-major, wT is C x K row-major
__global__ void wconv_kernel(const float* __restrict__ w, __bf16* __restrict__ wT,
                             int K, int C) {
  int idx = blockIdx.x * blockDim.x + threadIdx.x;
  if (idx >= K * C) return;
  int c = idx / K;
  int k = idx - c * K;
  wT[idx] = f2bf(w[k * C + c]);
}

// out_bf16 = bf16(a + b), elementwise
__global__ void add_bf16_kernel(const float* __restrict__ a, const float* __restrict__ b,
                                __bf16* __restrict__ out, int n) {
  int i = blockIdx.x * blockDim.x + threadIdx.x;
  if (i < n) out[i] = f2bf(a[i] + b[i]);
}

// Fused GINE message: for 16 edges per wave,
//   msg = relu( h[src] + edge_feats @ elW + elB ); atomicAdd(aggr[dst], msg)
// edge linear via v_wmma_f32_16x16x32_bf16; message tensor never hits HBM.
template <int C>
__global__ void edge_msg_kernel(const float* __restrict__ ef,     // E x 32 f32
                                const __bf16* __restrict__ wT,    // C x 32 bf16 (transposed)
                                const float* __restrict__ bias,   // C
                                const float* __restrict__ h,      // N x C f32 (L2-resident)
                                const int*  __restrict__ src,
                                const int*  __restrict__ dst,
                                float* __restrict__ aggr) {       // N x C f32
  int wave = blockIdx.x * (blockDim.x >> 5) + (threadIdx.x >> 5);
  int eb = wave * 16;
  if (eb >= NEDGES) return;                       // wave-uniform (EXEC all-1s for WMMA)
  int lane = threadIdx.x & 31;
  int m  = lane & 15;
  int kh = lane >> 4;

  // A fragment: 16 edges x 32 edge-features, converted f32 -> bf16 in-register.
  const float* ap = ef + (eb + m) * EDGE_DIM;
  v16bf a;
#pragma unroll
  for (int j = 0; j < 8; ++j) a[j]     = f2bf(ap[16 * kh + j]);       // K = 16kh .. 16kh+7
#pragma unroll
  for (int j = 0; j < 8; ++j) a[8 + j] = f2bf(ap[16 * kh + 8 + j]);   // K = 16kh+8 ..

  // Hoist row base pointers for the 8 output rows this half-wave owns.
  const float* hp[8];
  float*       gp[8];
#pragma unroll
  for (int r = 0; r < 8; ++r) {
    int e = eb + r + 8 * kh;
    hp[r] = h    + src[e] * C;
    gp[r] = aggr + dst[e] * C;
  }

#pragma unroll
  for (int t = 0; t < C / 16; ++t) {
    int col = t * 16 + m;
    // B fragment: W[16kh + j][col] == wT[col][16kh + j], contiguous 32B load
    v16bf b = *(const v16bf*)(wT + col * EDGE_DIM + 16 * kh);
    v8f c = {};
    c = __builtin_amdgcn_wmma_f32_16x16x32_bf16(false, a, false, b, (short)0, c,
                                                false, false);
    float bv = bias[col];
#pragma unroll
    for (int r = 0; r < 8; ++r) {
      float v = c[r] + bv + hp[r][col];            // gather: 64B coalesced per row
      v = v > 0.f ? v : 0.f;
      atomicAdd(gp[r] + col, v);                   // L2 atomic, 64B coalesced per row
    }
  }
}

// Dense GEMM: out[M x CO] = act[M x K](bf16) @ W[K x CO] + bias, optional ReLU,
// f32 and/or bf16 outputs. One wave per 16x64 output strip: 4 accumulators,
// 4 WMMAs per A-fragment load (4x less L2 read traffic on activations).
template <int K, int CO>
__global__ void gemm_kernel(const __bf16* __restrict__ A,    // M x K bf16
                            const __bf16* __restrict__ WT,   // CO x K bf16 (transposed)
                            const float* __restrict__ bias,  // CO
                            float* __restrict__ outF,        // M x CO or null
                            __bf16* __restrict__ outB,       // M x CO or null
                            int M, int relu) {
  constexpr int TN = 4;                            // 16x16 col tiles per wave
  constexpr int groups_n = CO / (16 * TN);
  int wave = blockIdx.x * (blockDim.x >> 5) + (threadIdx.x >> 5);
  int total = (M / 16) * groups_n;
  if (wave >= total) return;                       // wave-uniform
  int tm = wave / groups_n;
  int g  = wave - tm * groups_n;
  int lane = threadIdx.x & 31;
  int m  = lane & 15;
  int kh = lane >> 4;

  const __bf16* arow = A + (tm * 16 + m) * K;
  const __bf16* wrow[TN];
#pragma unroll
  for (int i = 0; i < TN; ++i) wrow[i] = WT + (g * 16 * TN + i * 16 + m) * K;

  v8f c[TN] = {};
#pragma unroll
  for (int ko = 0; ko < K; ko += 32) {
    v8bf lo = *(const v8bf*)(arow + ko + 16 * kh);       // K = ko+16kh .. +7
    v8bf hi = *(const v8bf*)(arow + ko + 16 * kh + 8);   // K = ko+16kh+8 ..
    v16bf a = __builtin_shufflevector(lo, hi, 0,1,2,3,4,5,6,7,8,9,10,11,12,13,14,15);
#pragma unroll
    for (int i = 0; i < TN; ++i) {
      v16bf b = *(const v16bf*)(wrow[i] + ko + 16 * kh); // K = ko+16kh .. +15
      c[i] = __builtin_amdgcn_wmma_f32_16x16x32_bf16(false, a, false, b, (short)0,
                                                     c[i], false, false);
    }
  }
#pragma unroll
  for (int i = 0; i < TN; ++i) {
    int col = g * 16 * TN + i * 16 + m;
    float bv = bias[col];
#pragma unroll
    for (int r = 0; r < 8; ++r) {
      int row = tm * 16 + r + 8 * kh;
      float v = c[i][r] + bv;
      if (relu) v = v > 0.f ? v : 0.f;
      if (outF) outF[row * CO + col] = v;
      if (outB) outB[row * CO + col] = f2bf(v);
    }
  }
}

extern "C" void kernel_launch(void* const* d_in, const int* in_sizes, int n_in,
                              void* d_out, int out_size, void* d_ws, size_t ws_size,
                              hipStream_t stream) {
  (void)in_sizes; (void)n_in; (void)out_size; (void)ws_size;
  const float* x   = (const float*)d_in[0];
  const int*   src = (const int*)d_in[1];
  const int*   dst = src + NEDGES;
  const float* ef  = (const float*)d_in[2];
  const float* el_w[3] = {(const float*)d_in[3], (const float*)d_in[9],  (const float*)d_in[15]};
  const float* el_b[3] = {(const float*)d_in[4], (const float*)d_in[10], (const float*)d_in[16]};
  const float* w1[3]   = {(const float*)d_in[5], (const float*)d_in[11], (const float*)d_in[17]};
  const float* b1[3]   = {(const float*)d_in[6], (const float*)d_in[12], (const float*)d_in[18]};
  const float* w2[3]   = {(const float*)d_in[7], (const float*)d_in[13], (const float*)d_in[19]};
  const float* b2[3]   = {(const float*)d_in[8], (const float*)d_in[14], (const float*)d_in[20]};
  const float* fc1_w = (const float*)d_in[21];
  const float* fc1_b = (const float*)d_in[22];
  const float* fc2_w = (const float*)d_in[23];
  const float* fc2_b = (const float*)d_in[24];

  char* ws = (char*)d_ws;
  size_t off = 0;
  auto walloc = [&](size_t bytes) -> void* {
    void* p = ws + off;
    off = (off + bytes + 255) & ~(size_t)255;
    return p;
  };
  float*  hA   = (float*) walloc((size_t)NNODES * HID * 4);
  float*  hB   = (float*) walloc((size_t)NNODES * HID * 4);
  float*  aggr = (float*) walloc((size_t)NNODES * HID * 4);
  __bf16* z0b  = (__bf16*)walloc((size_t)NNODES * HID * 2);
  __bf16* z1b  = (__bf16*)walloc((size_t)NNODES * HID * 2);
  int curdim[3] = {IN_DIM, HID, HID};
  __bf16 *elT[3], *w1T[3], *w2T[3];
  for (int l = 0; l < 3; ++l) {
    elT[l] = (__bf16*)walloc((size_t)curdim[l] * EDGE_DIM * 2);
    w1T[l] = (__bf16*)walloc((size_t)HID * curdim[l] * 2);
    w2T[l] = (__bf16*)walloc((size_t)HID * HID * 2);
  }
  __bf16* fc1T = (__bf16*)walloc((size_t)HID * HID * 2);
  __bf16* fc2T = (__bf16*)walloc((size_t)OUT_DIM * HID * 2);

  auto wconv = [&](const float* w, __bf16* wT, int K, int C) {
    int n = K * C;
    wconv_kernel<<<dim3((n + 255) / 256), dim3(256), 0, stream>>>(w, wT, K, C);
  };
  for (int l = 0; l < 3; ++l) {
    wconv(el_w[l], elT[l], EDGE_DIM, curdim[l]);
    wconv(w1[l],   w1T[l], curdim[l], HID);
    wconv(w2[l],   w2T[l], HID, HID);
  }
  wconv(fc1_w, fc1T, HID, HID);
  wconv(fc2_w, fc2T, HID, OUT_DIM);

  const dim3 eblk(128);                                  // 4 waves / block
  const dim3 egrd((NEDGES / 16 + 3) / 4);
  const int waves256 = (NNODES / 16) * (HID / 64);       // TN=4 -> CO/64 groups
  const dim3 g256((waves256 + 3) / 4);
  const int waves128 = (NNODES / 16) * (OUT_DIM / 64);
  const dim3 g128((waves128 + 3) / 4);

  // ----- layer 0 (C = 128, h = x) -----
  hipMemsetAsync(aggr, 0, (size_t)NNODES * IN_DIM * 4, stream);
  edge_msg_kernel<IN_DIM><<<egrd, eblk, 0, stream>>>(ef, elT[0], el_b[0], x, src, dst, aggr);
  {
    int n = NNODES * IN_DIM;
    add_bf16_kernel<<<dim3((n + 255) / 256), dim3(256), 0, stream>>>(x, aggr, z0b, n);
  }
  gemm_kernel<IN_DIM, HID><<<g256, eblk, 0, stream>>>(z0b, w1T[0], b1[0], nullptr, z1b, NNODES, 1);
  gemm_kernel<HID,    HID><<<g256, eblk, 0, stream>>>(z1b, w2T[0], b2[0], hA, nullptr, NNODES, 1);

  // ----- layers 1,2 (C = 256) -----
  float* hc = hA; float* hn = hB;
  for (int l = 1; l < 3; ++l) {
    hipMemsetAsync(aggr, 0, (size_t)NNODES * HID * 4, stream);
    edge_msg_kernel<HID><<<egrd, eblk, 0, stream>>>(ef, elT[l], el_b[l], hc, src, dst, aggr);
    int n = NNODES * HID;
    add_bf16_kernel<<<dim3((n + 255) / 256), dim3(256), 0, stream>>>(hc, aggr, z0b, n);
    gemm_kernel<HID, HID><<<g256, eblk, 0, stream>>>(z0b, w1T[l], b1[l], nullptr, z1b, NNODES, 1);
    if (l == 2) {  // final layer: only bf16(h3) needed downstream
      gemm_kernel<HID, HID><<<g256, eblk, 0, stream>>>(z1b, w2T[l], b2[l], nullptr, z0b, NNODES, 1);
    } else {
      gemm_kernel<HID, HID><<<g256, eblk, 0, stream>>>(z1b, w2T[l], b2[l], hn, nullptr, NNODES, 1);
    }
    float* t = hc; hc = hn; hn = t;
  }

  // ----- head: relu(h3 @ fc1 + b) @ fc2 + b -> d_out (f32) -----
  gemm_kernel<HID, HID>    <<<g256, eblk, 0, stream>>>(z0b, fc1T, fc1_b, nullptr, z1b, NNODES, 1);
  gemm_kernel<HID, OUT_DIM><<<g128, eblk, 0, stream>>>(z1b, fc2T, fc2_b, (float*)d_out, nullptr, NNODES, 0);
}
